// WaveglowBlock_66675072303332
// MI455X (gfx1250) — matrix operand order repaired and checked by reading, not compiled, not run on hardware
//
#include <hip/hip_runtime.h>
#include <hip/hip_bf16.h>

// ---------------------------------------------------------------------------
// WaveGlow WN block for MI455X (gfx1250), wave32 + v_wmma_f32_16x16x32_f16.
// Double-buffered LDS A tiles; async global->LDS staging via
// GLOBAL_LOAD_ASYNC_TO_LDS_B128 (ASYNCcnt path), else load+ds_store fallback.
// ---------------------------------------------------------------------------

typedef _Float16 v16h __attribute__((ext_vector_type(16)));
typedef _Float16 v8h  __attribute__((ext_vector_type(8)));
typedef float    v8f  __attribute__((ext_vector_type(8)));
typedef int      v4i  __attribute__((ext_vector_type(4)));

#define BT      65536
#define TLEN    8192
#define NCH     256
#define NCH2    512
#define NMEL    640
#define LDSW    40        // padded LDS row stride (80B: 16B-aligned, bank-spread)

#if defined(__gfx1250__) && __has_builtin(__builtin_amdgcn_global_load_async_to_lds_b128)
#define HAVE_ASYNC_LDS 1
typedef __attribute__((address_space(1))) v4i* as1_v4i_p;   // global int4*
typedef __attribute__((address_space(3))) v4i* as3_v4i_p;   // LDS int4*
#else
#define HAVE_ASYNC_LDS 0
#endif

__device__ __forceinline__ void wait_async0() {
#if defined(__gfx1250__)
#if __has_builtin(__builtin_amdgcn_s_wait_asynccnt)
    __builtin_amdgcn_s_wait_asynccnt(0);
#else
    asm volatile("s_wait_asynccnt 0x0" ::: "memory");
#endif
#endif
}

// stage 16 bytes global -> LDS (zero-fill if !ok)
__device__ __forceinline__ void stage16(const _Float16* gp, _Float16* lp, int ok) {
#if HAVE_ASYNC_LDS
    if (ok) {
        __builtin_amdgcn_global_load_async_to_lds_b128(
            (as1_v4i_p)gp, (as3_v4i_p)lp, 0, 0);
    } else {
        v8h z;
#pragma unroll
        for (int i = 0; i < 8; ++i) z[i] = (_Float16)0.f;
        *(v8h*)lp = z;
    }
#else
    v8h v;
#pragma unroll
    for (int i = 0; i < 8; ++i) v[i] = (_Float16)0.f;
    if (ok) v = *(const v8h*)gp;
    *(v8h*)lp = v;
#endif
}

// ---- fragment helpers ------------------------------------------------------

// A fragment (16x32 f16) from LDS-staged row-major tile.
__device__ __forceinline__ v16h frag_a_lds(const _Float16* lds, int rowBase, int lane) {
    int r  = rowBase + (lane & 15);
    int kk = (lane & 16) ? 16 : 0;
    const _Float16* p = lds + r * LDSW + kk;
    v8h lo = *(const v8h*)(p);
    v8h hi = *(const v8h*)(p + 8);
    v16h o;
#pragma unroll
    for (int i = 0; i < 8; ++i) { o[i] = lo[i]; o[i + 8] = hi[i]; }
    return o;
}

// B fragment (32x16 f16) from pre-packed memory: frag base + lane*16 halves.
__device__ __forceinline__ v16h frag_b_glb(const _Float16* fragBase, int lane) {
    const _Float16* p = fragBase + lane * 16;
    v8h lo = *(const v8h*)(p);
    v8h hi = *(const v8h*)(p + 8);
    v16h o;
#pragma unroll
    for (int i = 0; i < 8; ++i) { o[i] = lo[i]; o[i + 8] = hi[i]; }
    return o;
}

__device__ __forceinline__ v8f zero8f() {
    v8f z;
#pragma unroll
    for (int i = 0; i < 8; ++i) z[i] = 0.0f;
    return z;
}

// ---- weight repack: W[K][N] fp32 row-major -> WMMA B-fragment order --------
// lane<16 : W[kt*32 +    j][nt*16 + lane]      (j = 0..15)
// lane>=16: W[kt*32 + 16+j][nt*16 + lane-16]
__global__ void pack_b_kernel(const float* __restrict__ W, _Float16* __restrict__ out,
                              int K, int N) {
    int idx = blockIdx.x * 256 + threadIdx.x;
    int nFragN = N >> 4;
    int total  = (K >> 5) * nFragN * 32;
    if (idx >= total) return;
    int lane = idx & 31;
    int frag = idx >> 5;
    int nt = frag % nFragN;
    int kt = frag / nFragN;
    int n  = (nt << 4) + (lane & 15);
    int kb = (kt << 5) + ((lane & 16) ? 16 : 0);
    _Float16* dst = out + (size_t)idx * 16;
#pragma unroll
    for (int j = 0; j < 16; ++j)
        dst[j] = (_Float16)W[(size_t)(kb + j) * N + n];
}

// ---- fp32 -> f16 conversion (spect staging) --------------------------------
__global__ void cvt_f16_kernel(const float* __restrict__ src, _Float16* __restrict__ dst,
                               unsigned long long n) {
    unsigned long long i = (unsigned long long)blockIdx.x * 256ull + threadIdx.x;
    if (i < n) dst[i] = (_Float16)src[i];
}

// ---- start: h = audio @ start_w + start_b  (K=4, trivial) ------------------
__global__ void start_kernel(const float* __restrict__ audio, const float* __restrict__ w,
                             const float* __restrict__ b,
                             float* __restrict__ h32, _Float16* __restrict__ h16) {
    int idx = blockIdx.x * 256 + threadIdx.x;
    int c = idx & 255;
    int t = idx >> 8;
    const float* a = audio + (size_t)t * 4;
    float s = b[c];
#pragma unroll
    for (int j = 0; j < 4; ++j) s += a[j] * w[j * NCH + c];
    h32[idx] = s;
    h16[idx] = (_Float16)s;
}

// ---- GEMM1: x = [h(t-d),h(t),h(t+d),spect] @ Wcat + biases; gated act ------
// 256 threads (8 waves), 64 rows x 512 cols, K=1408 (44 k-tiles), double-buf.
__global__ void __launch_bounds__(256)
gemm1_kernel(const _Float16* __restrict__ h16, const _Float16* __restrict__ sp16,
             const _Float16* __restrict__ wcat,
             const float* __restrict__ in_b, const float* __restrict__ cond_b,
             _Float16* __restrict__ acts, int dilation) {
    __shared__ _Float16 la[2][64 * LDSW];
    const int tid  = threadIdx.x;
    const int wave = tid >> 5;
    const int lane = tid & 31;
    const int row0 = blockIdx.x * 64;

    v8f accT[4][2], accS[4][2];
#pragma unroll
    for (int rt = 0; rt < 4; ++rt)
#pragma unroll
        for (int j = 0; j < 2; ++j) { accT[rt][j] = zero8f(); accS[rt][j] = zero8f(); }

    const int fr = tid >> 2;            // fill row 0..63
    const int fk = (tid & 3) * 8;       // fill k-chunk 0/8/16/24
    const int grow  = row0 + fr;
    const int batch = grow & ~(TLEN - 1);
    const int tloc  = grow & (TLEN - 1);
    const _Float16* spsrc = sp16 + (size_t)grow * NMEL + fk;

    auto fill = [&](int kt, int buf) {
        _Float16* lp = &la[buf][fr * LDSW + fk];
        if (kt < 24) {
            int tap = kt >> 3;
            int ts  = tloc + (tap - 1) * dilation;
            int ok  = ((unsigned)ts < (unsigned)TLEN);
            const _Float16* gp =
                h16 + (size_t)(batch + (ok ? ts : 0)) * NCH + (kt & 7) * 32 + fk;
            stage16(gp, lp, ok);
        } else {
            stage16(spsrc + (kt - 24) * 32, lp, 1);
        }
    };

    fill(0, 0);
    for (int kt = 0; kt < 44; ++kt) {
#if HAVE_ASYNC_LDS
        wait_async0();
#endif
        __syncthreads();
        if (kt < 43) fill(kt + 1, (kt + 1) & 1);   // overlaps with WMMAs below

        const _Float16* lbuf = la[kt & 1];
        const _Float16* bbase = wcat + (size_t)kt * 32 * 512;
        __builtin_prefetch(bbase + 32 * 512 + (size_t)wave * 2 * 512, 0, 1);

        v16h af[4];
#pragma unroll
        for (int rt = 0; rt < 4; ++rt) af[rt] = frag_a_lds(lbuf, rt * 16, lane);
#pragma unroll
        for (int j = 0; j < 2; ++j) {
            v16h bT = frag_b_glb(bbase + (size_t)(wave * 2 + j) * 512, lane);
            v16h bS = frag_b_glb(bbase + (size_t)(16 + wave * 2 + j) * 512, lane);
#pragma unroll
            for (int rt = 0; rt < 4; ++rt) {
                accT[rt][j] = __builtin_amdgcn_wmma_f32_16x16x32_f16(
                    false, af[rt], false, bT, (short)0, accT[rt][j], false, false);
                accS[rt][j] = __builtin_amdgcn_wmma_f32_16x16x32_f16(
                    false, af[rt], false, bS, (short)0, accS[rt][j], false, false);
            }
        }
    }

    // gated activation: acts[c] = tanh(x[c]) * sigmoid(x[c+256])
#pragma unroll
    for (int j = 0; j < 2; ++j) {
        int col = wave * 32 + j * 16 + (lane & 15);
        float bT = in_b[col] + cond_b[col];
        float bS = in_b[col + NCH] + cond_b[col + NCH];
#pragma unroll
        for (int rt = 0; rt < 4; ++rt)
#pragma unroll
            for (int v = 0; v < 8; ++v) {
                float xt = accT[rt][j][v] + bT;
                float xs = accS[rt][j][v] + bS;
                float a  = tanhf(xt) * (1.0f / (1.0f + __expf(-xs)));
                int r = row0 + rt * 16 + v + ((lane >> 4) << 3);
                acts[(size_t)r * NCH + col] = (_Float16)a;
            }
    }
}

// ---- GEMM2: rs = acts @ res_skip_w; h += rs[:,:256]; skip (+)= rs[:,256:] --
__global__ void __launch_bounds__(256)
gemm2_kernel(const _Float16* __restrict__ acts, const _Float16* __restrict__ wrs,
             const float* __restrict__ rs_b, float* __restrict__ h32,
             _Float16* __restrict__ h16, float* __restrict__ skip, int first) {
    __shared__ _Float16 la[2][64 * LDSW];
    const int tid  = threadIdx.x;
    const int wave = tid >> 5;
    const int lane = tid & 31;
    const int row0 = blockIdx.x * 64;

    v8f accT[4][2], accS[4][2];
#pragma unroll
    for (int rt = 0; rt < 4; ++rt)
#pragma unroll
        for (int j = 0; j < 2; ++j) { accT[rt][j] = zero8f(); accS[rt][j] = zero8f(); }

    const int fr = tid >> 2;
    const int fk = (tid & 3) * 8;
    const int grow = row0 + fr;
    const _Float16* asrc = acts + (size_t)grow * NCH + fk;

    auto fill = [&](int kt, int buf) {
        stage16(asrc + kt * 32, &la[buf][fr * LDSW + fk], 1);
    };

    fill(0, 0);
    for (int kt = 0; kt < 8; ++kt) {
#if HAVE_ASYNC_LDS
        wait_async0();
#endif
        __syncthreads();
        if (kt < 7) fill(kt + 1, (kt + 1) & 1);

        const _Float16* lbuf = la[kt & 1];
        const _Float16* bbase = wrs + (size_t)kt * 32 * 512;
        __builtin_prefetch(bbase + 32 * 512 + (size_t)wave * 2 * 512, 0, 1);

        v16h af[4];
#pragma unroll
        for (int rt = 0; rt < 4; ++rt) af[rt] = frag_a_lds(lbuf, rt * 16, lane);
#pragma unroll
        for (int j = 0; j < 2; ++j) {
            v16h bT = frag_b_glb(bbase + (size_t)(wave * 2 + j) * 512, lane);
            v16h bS = frag_b_glb(bbase + (size_t)(16 + wave * 2 + j) * 512, lane);
#pragma unroll
            for (int rt = 0; rt < 4; ++rt) {
                accT[rt][j] = __builtin_amdgcn_wmma_f32_16x16x32_f16(
                    false, af[rt], false, bT, (short)0, accT[rt][j], false, false);
                accS[rt][j] = __builtin_amdgcn_wmma_f32_16x16x32_f16(
                    false, af[rt], false, bS, (short)0, accS[rt][j], false, false);
            }
        }
    }

#pragma unroll
    for (int j = 0; j < 2; ++j) {
        int col = wave * 32 + j * 16 + (lane & 15);
        float bT = rs_b[col];
        float bS = rs_b[col + NCH];
#pragma unroll
        for (int rt = 0; rt < 4; ++rt)
#pragma unroll
            for (int v = 0; v < 8; ++v) {
                int r = row0 + rt * 16 + v + ((lane >> 4) << 3);
                size_t idx = (size_t)r * NCH + col;
                float hn = accT[rt][j][v] + bT + h32[idx];   // residual in fp32
                h32[idx] = hn;
                h16[idx] = (_Float16)hn;
                float sk = accS[rt][j][v] + bS;
                skip[idx] = first ? sk : (skip[idx] + sk);
            }
    }
}

// ---- last layer: skip += acts @ res_last_w + res_last_b  (N=256) -----------
__global__ void __launch_bounds__(256)
gemm2_last_kernel(const _Float16* __restrict__ acts, const _Float16* __restrict__ wl,
                  const float* __restrict__ rl_b, float* __restrict__ skip) {
    __shared__ _Float16 la[2][64 * LDSW];
    const int tid  = threadIdx.x;
    const int wave = tid >> 5;
    const int lane = tid & 31;
    const int row0 = blockIdx.x * 64;

    v8f acc[4][2];
#pragma unroll
    for (int rt = 0; rt < 4; ++rt)
#pragma unroll
        for (int j = 0; j < 2; ++j) acc[rt][j] = zero8f();

    const int fr = tid >> 2;
    const int fk = (tid & 3) * 8;
    const int grow = row0 + fr;
    const _Float16* asrc = acts + (size_t)grow * NCH + fk;

    auto fill = [&](int kt, int buf) {
        stage16(asrc + kt * 32, &la[buf][fr * LDSW + fk], 1);
    };

    fill(0, 0);
    for (int kt = 0; kt < 8; ++kt) {
#if HAVE_ASYNC_LDS
        wait_async0();
#endif
        __syncthreads();
        if (kt < 7) fill(kt + 1, (kt + 1) & 1);

        const _Float16* lbuf = la[kt & 1];
        const _Float16* bbase = wl + (size_t)kt * 16 * 512;  // nFragN = 16
        v16h af[4];
#pragma unroll
        for (int rt = 0; rt < 4; ++rt) af[rt] = frag_a_lds(lbuf, rt * 16, lane);
#pragma unroll
        for (int j = 0; j < 2; ++j) {
            v16h b = frag_b_glb(bbase + (size_t)(wave * 2 + j) * 512, lane);
#pragma unroll
            for (int rt = 0; rt < 4; ++rt)
                acc[rt][j] = __builtin_amdgcn_wmma_f32_16x16x32_f16(
                    false, af[rt], false, b, (short)0, acc[rt][j], false, false);
        }
    }

#pragma unroll
    for (int j = 0; j < 2; ++j) {
        int col = wave * 32 + j * 16 + (lane & 15);
        float bb = rl_b[col];
#pragma unroll
        for (int rt = 0; rt < 4; ++rt)
#pragma unroll
            for (int v = 0; v < 8; ++v) {
                int r = row0 + rt * 16 + v + ((lane >> 4) << 3);
                size_t idx = (size_t)r * NCH + col;
                skip[idx] += acc[rt][j][v] + bb;
            }
    }
}

// ---- end: out = skip @ end_w + end_b  (N=8, trivial) -----------------------
__global__ void end_kernel(const float* __restrict__ skip, const float* __restrict__ w,
                           const float* __restrict__ b, float* __restrict__ out) {
    int idx = blockIdx.x * 256 + threadIdx.x;
    if (idx >= BT * 8) return;
    int o = idx & 7;
    int t = idx >> 3;
    const float* sp = skip + (size_t)t * NCH;
    float s = b[o];
    for (int c = 0; c < NCH; ++c) s += sp[c] * w[c * 8 + o];
    out[idx] = s;
}

// ---------------------------------------------------------------------------
extern "C" void kernel_launch(void* const* d_in, const int* in_sizes, int n_in,
                              void* d_out, int out_size, void* d_ws, size_t ws_size,
                              hipStream_t stream) {
    (void)in_sizes; (void)n_in; (void)out_size; (void)ws_size;
    const float* audio   = (const float*)d_in[0];
    const float* spect   = (const float*)d_in[1];
    const float* start_w = (const float*)d_in[2];
    const float* start_b = (const float*)d_in[3];
    const float* in_w    = (const float*)d_in[4];   // (8,3,256,512)
    const float* in_b    = (const float*)d_in[5];   // (8,512)
    const float* cond_w  = (const float*)d_in[6];   // (8,640,512)
    const float* cond_b  = (const float*)d_in[7];   // (8,512)
    const float* rs_w    = (const float*)d_in[8];   // (7,256,512)
    const float* rs_b    = (const float*)d_in[9];   // (7,512)
    const float* rl_w    = (const float*)d_in[10];  // (256,256)
    const float* rl_b    = (const float*)d_in[11];  // (256,)
    const float* end_w   = (const float*)d_in[12];  // (256,8)
    const float* end_b   = (const float*)d_in[13];  // (8,)
    float* out = (float*)d_out;

    char* p = (char*)d_ws;
    float*     h32  = (float*)p;      p += (size_t)BT * NCH * 4;
    float*     skip = (float*)p;      p += (size_t)BT * NCH * 4;
    _Float16*  h16  = (_Float16*)p;   p += (size_t)BT * NCH * 2;
    _Float16*  acts = (_Float16*)p;   p += (size_t)BT * NCH * 2;
    _Float16*  sp16 = (_Float16*)p;   p += (size_t)BT * NMEL * 2;
    const size_t WCAT_L = (size_t)44 * 32 * 512;
    _Float16*  wcat = (_Float16*)p;   p += 8 * WCAT_L * 2;
    const size_t WRS_L = (size_t)8 * 32 * 512;
    _Float16*  wrs  = (_Float16*)p;   p += 7 * WRS_L * 2;
    _Float16*  wlast = (_Float16*)p;  p += (size_t)8 * 16 * 512 * 2;

    {
        unsigned long long n = (unsigned long long)BT * NMEL;
        cvt_f16_kernel<<<(unsigned)((n + 255) / 256), 256, 0, stream>>>(spect, sp16, n);
    }
    for (int l = 0; l < 8; ++l) {
        _Float16* wl = wcat + (size_t)l * WCAT_L;
        pack_b_kernel<<<(24 * 32 * 32 + 255) / 256, 256, 0, stream>>>(
            in_w + (size_t)l * 3 * NCH * NCH2, wl, 768, 512);
        pack_b_kernel<<<(20 * 32 * 32 + 255) / 256, 256, 0, stream>>>(
            cond_w + (size_t)l * NMEL * NCH2, wl + (size_t)24 * 32 * 512, 640, 512);
    }
    for (int l = 0; l < 7; ++l)
        pack_b_kernel<<<(8 * 32 * 32 + 255) / 256, 256, 0, stream>>>(
            rs_w + (size_t)l * NCH * NCH2, wrs + (size_t)l * WRS_L, 256, 512);
    pack_b_kernel<<<(8 * 16 * 32 + 255) / 256, 256, 0, stream>>>(rl_w, wlast, 256, 256);

    start_kernel<<<BT * NCH / 256, 256, 0, stream>>>(audio, start_w, start_b, h32, h16);

    for (int l = 0; l < 8; ++l) {
        gemm1_kernel<<<BT / 64, 256, 0, stream>>>(
            h16, sp16, wcat + (size_t)l * WCAT_L,
            in_b + (size_t)l * NCH2, cond_b + (size_t)l * NCH2, acts, 1 << l);
        if (l < 7)
            gemm2_kernel<<<BT / 64, 256, 0, stream>>>(
                acts, wrs + (size_t)l * WRS_L, rs_b + (size_t)l * NCH2,
                h32, h16, skip, l == 0 ? 1 : 0);
        else
            gemm2_last_kernel<<<BT / 64, 256, 0, stream>>>(acts, wlast, rl_b, skip);
    }

    end_kernel<<<BT * 8 / 256, 256, 0, stream>>>(skip, end_w, end_b, out);
}
